// REGNN_66073776881864
// MI455X (gfx1250) — matrix operand order repaired and checked by reading, not compiled, not run on hardware
//
#include <hip/hip_runtime.h>
#include <hip/hip_bf16.h>
#include <stdint.h>

typedef __attribute__((ext_vector_type(16))) _Float16 v16h;
typedef __attribute__((ext_vector_type(8)))  float    v8f;
typedef __attribute__((ext_vector_type(4)))  float    f4v;
typedef __attribute__((ext_vector_type(4)))  _Float16 h4v;

#define PER_TYPE 250000
#define NTYPES   4
#define HDIM     128
#define OUTC     349
#define OUTP     352
#define NN0      400000
#define NN1      100000
#define NN2      25000
#define EE0      1500000
#define EE1      375000

// ---------------------------------------------------------------------------
// WMMA helper: D = A(16x32 f16) * B(32x16 f16) + C(16x16 f32)
// ---------------------------------------------------------------------------
static __device__ __forceinline__ v8f wmma16(v16h a, v16h b, v8f c) {
    return __builtin_amdgcn_wmma_f32_16x16x32_f16(
        /*neg_a=*/false, a, /*neg_b=*/false, b,
        /*c_mod=*/(short)0, c, /*reuse_a=*/false, /*reuse_b=*/false);
}

// Build the A fragment for K-chunk c (K = 32c..32c+31) for this lane.
// ISA 16-bit A 16x32 layout: element e of v16h holds K = 32c + 16*(e/8) + 8*h + (e%8),
// where h = lane/16 and the lane's row is m = lane%16.
static __device__ __forceinline__ v16h load_a_frag(const float* __restrict__ row, int c, int h) {
    const float* q = row + 32 * c + 8 * h;
    f4v g0 = *(const f4v*)(q + 0);
    f4v g1 = *(const f4v*)(q + 4);
    f4v g2 = *(const f4v*)(q + 16);
    f4v g3 = *(const f4v*)(q + 20);
    v16h a;
#pragma unroll
    for (int i = 0; i < 4; ++i) {
        a[i]      = (_Float16)g0[i];
        a[4 + i]  = (_Float16)g1[i];
        a[8 + i]  = (_Float16)g2[i];
        a[12 + i] = (_Float16)g3[i];
    }
    return a;
}

// ---------------------------------------------------------------------------
// Weight pre-pack: W[mat][K=128][ncols] (f32) -> f16 fragments laid out so each
// lane's B fragment is 16 contiguous halves:
//   dst[(((mat*njt + j)*4 + c)*32 + lane)*16 + e]
//     = (f16) W[mat][32c + 16*(e/8) + 8*(lane/16) + (e%8)][j*16 + lane%16]
// (zero-padded for col >= ncols)
// ---------------------------------------------------------------------------
__global__ void k_pack(const float* __restrict__ W, _Float16* __restrict__ dst,
                       int nmat, int njt, int ncols) {
    long long total = (long long)nmat * njt * 4 * 32 * 16;
    long long id = (long long)blockIdx.x * blockDim.x + threadIdx.x;
    if (id >= total) return;
    int e = (int)(id & 15);
    int l = (int)((id >> 4) & 31);
    int c = (int)((id >> 9) & 3);
    long long rest = id >> 11;
    int j   = (int)(rest % njt);
    int mat = (int)(rest / njt);
    int K   = 32 * c + 16 * (e >> 3) + 8 * (l >> 4) + (e & 7);
    int col = j * 16 + (l & 15);
    float v = (col < ncols) ? W[((long long)mat * 128 + K) * ncols + col] : 0.0f;
    dst[id] = (_Float16)v;
}

// ---------------------------------------------------------------------------
// Input projection: x[i] = feats[type(i), local(i)] @ W_in[type(i)] + b_in[type(i)]
// One block = one 16-row tile, 8 waves = 8 col tiles of 16. Mixed types inside a
// tile handled by masked accumulation over the 4 type weight matrices (the mask
// is a per-lane register select; EXEC is all-ones at every WMMA).
// ---------------------------------------------------------------------------
__global__ void k_input_proj(const float* __restrict__ feats,
                             const int* __restrict__ n_id,
                             const int* __restrict__ node_type,
                             const int* __restrict__ local_idx,
                             const _Float16* __restrict__ WinPack, // [4][8][4][32][16]
                             const float* __restrict__ b_in,       // [4][128]
                             float* __restrict__ xout)             // [N0][128]
{
    const int wave = threadIdx.x >> 5;   // col tile j (0..7)
    const int lane = threadIdx.x & 31;
    const int m = lane & 15;
    const int h = lane >> 4;
    const int tile0 = blockIdx.x * 16;

    const int gid = n_id[tile0 + m];
    const int rt  = node_type[gid];
    const int li  = local_idx[gid];
    const float* rowp = feats + ((long long)rt * PER_TYPE + li) * HDIM;

    v16h a[4];
#pragma unroll
    for (int c = 0; c < 4; ++c) a[c] = load_a_frag(rowp, c, h);

    const v16h zero = {};
    v8f acc = {};
    const int j = wave;
#pragma unroll
    for (int t = 0; t < NTYPES; ++t) {
        const _Float16* bp = WinPack + (((long long)t * 8 + j) * 4) * 512 + lane * 16;
        // Preload all 4 K-chunk B fragments, then run the WMMA chain back-to-back.
        v16h b0 = *(const v16h*)(bp + 0 * 512);
        v16h b1 = *(const v16h*)(bp + 1 * 512);
        v16h b2 = *(const v16h*)(bp + 2 * 512);
        v16h b3 = *(const v16h*)(bp + 3 * 512);
        v16h a0 = (rt == t) ? a[0] : zero;
        v16h a1 = (rt == t) ? a[1] : zero;
        v16h a2 = (rt == t) ? a[2] : zero;
        v16h a3 = (rt == t) ? a[3] : zero;
        acc = wmma16(a0, b0, acc);
        acc = wmma16(a1, b1, acc);
        acc = wmma16(a2, b2, acc);
        acc = wmma16(a3, b3, acc);
    }

    const int col = j * 16 + (lane & 15);
#pragma unroll
    for (int r = 0; r < 8; ++r) {
        int grow = tile0 + r + 8 * h;                  // C/D layout: M = r + 8*(lane/16)
        int t2 = node_type[n_id[grow]];
        xout[(long long)grow * HDIM + col] = acc[r] + b_in[t2 * HDIM + col];
    }
}

// ---------------------------------------------------------------------------
// Generic GEMM: out = X[nrows x 128] @ Wpack + bias ; out f16 (hh) or f32 (logits)
// ---------------------------------------------------------------------------
template <bool OUT_F16>
__global__ void k_gemm(const float* __restrict__ X,
                       const _Float16* __restrict__ Bpack,
                       const float* __restrict__ bias,
                       void* __restrict__ outv,
                       int nrows, int njt, int ncols, int ldout)
{
    const int wave = threadIdx.x >> 5;
    const int lane = threadIdx.x & 31;
    const int m = lane & 15;
    const int h = lane >> 4;
    const int tile0 = blockIdx.x * 16;

    int rowc = tile0 + m;
    if (rowc >= nrows) rowc = nrows - 1;               // clamp loads for partial tiles
    const float* rowp = X + (long long)rowc * HDIM;

    v16h a[4];
#pragma unroll
    for (int c = 0; c < 4; ++c) a[c] = load_a_frag(rowp, c, h);

    for (int j = wave; j < njt; j += 8) {
        const _Float16* bp = Bpack + ((long long)j * 4) * 512 + lane * 16;
        v16h b0 = *(const v16h*)(bp + 0 * 512);
        v16h b1 = *(const v16h*)(bp + 1 * 512);
        v16h b2 = *(const v16h*)(bp + 2 * 512);
        v16h b3 = *(const v16h*)(bp + 3 * 512);
        v8f acc = {};
        acc = wmma16(a[0], b0, acc);
        acc = wmma16(a[1], b1, acc);
        acc = wmma16(a[2], b2, acc);
        acc = wmma16(a[3], b3, acc);

        const int col = j * 16 + (lane & 15);
        if (col < ncols) {
            const float bb = bias[col];
#pragma unroll
            for (int r = 0; r < 8; ++r) {
                int grow = tile0 + r + 8 * h;
                if (grow < nrows) {
                    float v = acc[r] + bb;
                    if (OUT_F16)
                        ((_Float16*)outv)[(long long)grow * ldout + col] = (_Float16)v;
                    else
                        ((float*)outv)[(long long)grow * ldout + col] = v;
                }
            }
        }
    }
}

// ---------------------------------------------------------------------------
// Zero a float buffer
// ---------------------------------------------------------------------------
__global__ void k_zero(float* __restrict__ p, long long n) {
    long long i = (long long)blockIdx.x * blockDim.x + threadIdx.x;
    long long step = (long long)gridDim.x * blockDim.x;
    for (; i < n; i += step) p[i] = 0.0f;
}

// ---------------------------------------------------------------------------
// Edge scatter: agg[dst] += rel_w[et] * hh[src].
// Persistent waves, one edge per wave per iteration, 4 cols/lane.
// Prefetch the next edge's hh row to hide the random 256B gather latency.
// ---------------------------------------------------------------------------
__global__ void k_scatter(const _Float16* __restrict__ hh,
                          const int* __restrict__ src,
                          const int* __restrict__ dst,
                          const int* __restrict__ et,
                          const float* __restrict__ rw,   // 8 relation weights
                          float* __restrict__ agg,
                          int ne)
{
    const int lane = (int)(threadIdx.x & 31);
    long long wid = ((long long)blockIdx.x * blockDim.x + threadIdx.x) >> 5;
    const long long nw = ((long long)gridDim.x * blockDim.x) >> 5;

    for (long long e = wid; e < ne; e += nw) {
        long long en = e + nw;
        if (en < ne) {
            int sn = src[en];
            __builtin_prefetch(hh + (long long)sn * HDIM + lane * 4, 0, 1);
        }
        const int s = src[e];
        const int d = dst[e];
        const float w = rw[et[e]];
        h4v v = *(const h4v*)(hh + (long long)s * HDIM + lane * 4);
        float* ap = agg + (long long)d * HDIM + lane * 4;
        atomicAdd(ap + 0, w * (float)v[0]);
        atomicAdd(ap + 1, w * (float)v[1]);
        atomicAdd(ap + 2, w * (float)v[2]);
        atomicAdd(ap + 3, w * (float)v[3]);
    }
}

// ---------------------------------------------------------------------------
// Combine: xout = relu(agg + self_w[type]*hh[:ntgt] + xin[:ntgt])
// ---------------------------------------------------------------------------
__global__ void k_combine(const float* __restrict__ agg,
                          const _Float16* __restrict__ hh,
                          const float* __restrict__ xin,
                          const int* __restrict__ n_id,
                          const int* __restrict__ node_type,
                          const float* __restrict__ sw,   // 4 self weights
                          float* __restrict__ xout,
                          int ntgt)
{
    long long tid = (long long)blockIdx.x * blockDim.x + threadIdx.x;
    if (tid >= (long long)ntgt * HDIM) return;
    long long i = tid >> 7;
    int t = node_type[n_id[i]];
    float v = agg[tid] + sw[t] * (float)hh[tid] + xin[tid];
    xout[tid] = v > 0.0f ? v : 0.0f;
}

// ---------------------------------------------------------------------------
// Row-wise log-softmax over 349 logits (stride OUTP), one block of 128 per row
// ---------------------------------------------------------------------------
__global__ void k_lsm(const float* __restrict__ logits, float* __restrict__ out) {
    __shared__ float red[128];
    const int row = blockIdx.x;
    const float* lp = logits + (long long)row * OUTP;

    float mx = -3.4e38f;
    for (int c = threadIdx.x; c < OUTC; c += 128) mx = fmaxf(mx, lp[c]);
    red[threadIdx.x] = mx;
    __syncthreads();
    for (int s = 64; s > 0; s >>= 1) {
        if (threadIdx.x < s) red[threadIdx.x] = fmaxf(red[threadIdx.x], red[threadIdx.x + s]);
        __syncthreads();
    }
    mx = red[0];
    __syncthreads();

    float sum = 0.0f;
    for (int c = threadIdx.x; c < OUTC; c += 128) sum += expf(lp[c] - mx);
    red[threadIdx.x] = sum;
    __syncthreads();
    for (int s = 64; s > 0; s >>= 1) {
        if (threadIdx.x < s) red[threadIdx.x] += red[threadIdx.x + s];
        __syncthreads();
    }
    const float lse = logf(red[0]) + mx;

    for (int c = threadIdx.x; c < OUTC; c += 128)
        out[(long long)row * OUTC + c] = lp[c] - lse;
}

// ---------------------------------------------------------------------------
// Host launcher
// ---------------------------------------------------------------------------
extern "C" void kernel_launch(void* const* d_in, const int* in_sizes, int n_in,
                              void* d_out, int out_size, void* d_ws, size_t ws_size,
                              hipStream_t stream) {
    (void)in_sizes; (void)n_in; (void)out_size; (void)ws_size;

    const float* feats      = (const float*)d_in[0];
    const int*   n_id       = (const int*)d_in[1];
    const int*   node_type  = (const int*)d_in[2];
    const int*   local_idx  = (const int*)d_in[3];
    const int*   edge_src0  = (const int*)d_in[4];
    const int*   edge_dst0  = (const int*)d_in[5];
    const int*   etype0     = (const int*)d_in[6];
    const int*   edge_src1  = (const int*)d_in[7];
    const int*   edge_dst1  = (const int*)d_in[8];
    const int*   etype1     = (const int*)d_in[9];
    const float* W_in       = (const float*)d_in[10];
    const float* b_in       = (const float*)d_in[11];
    const float* W_conv     = (const float*)d_in[12];
    const float* b_conv     = (const float*)d_in[13];
    const float* rel_w      = (const float*)d_in[14];
    const float* self_w     = (const float*)d_in[15];
    const float* W_out      = (const float*)d_in[16];
    const float* b_out      = (const float*)d_in[17];
    float* out = (float*)d_out;

    // Workspace carve-up (256B aligned)
    char* ws = (char*)d_ws;
    size_t off = 0;
    auto take = [&](size_t bytes) -> char* {
        char* p = ws + off;
        off += (bytes + 255) & ~(size_t)255;
        return p;
    };
    float*    x0     = (float*)   take((size_t)NN0 * HDIM * 4);   // 204.8 MB
    _Float16* hh     = (_Float16*)take((size_t)NN0 * HDIM * 2);   // 102.4 MB (reused both layers)
    float*    agg    = (float*)   take((size_t)NN1 * HDIM * 4);   //  51.2 MB (reused both layers)
    float*    x1     = (float*)   take((size_t)NN1 * HDIM * 4);   //  51.2 MB
    float*    x2     = (float*)   take((size_t)NN2 * HDIM * 4);   //  12.8 MB
    float*    logits = (float*)   take((size_t)NN2 * OUTP * 4);   //  35.2 MB
    _Float16* pWin   = (_Float16*)take((size_t)NTYPES * 8 * 4 * 32 * 16 * 2);
    _Float16* pWc    = (_Float16*)take((size_t)2 * 8 * 4 * 32 * 16 * 2);
    _Float16* pWo    = (_Float16*)take((size_t)1 * 22 * 4 * 32 * 16 * 2);

    // --- pack weights into f16 B-fragment layout ---
    {
        int tot = NTYPES * 8 * 2048;
        k_pack<<<(tot + 255) / 256, 256, 0, stream>>>(W_in, pWin, NTYPES, 8, HDIM);
        tot = 2 * 8 * 2048;
        k_pack<<<(tot + 255) / 256, 256, 0, stream>>>(W_conv, pWc, 2, 8, HDIM);
        tot = 1 * 22 * 2048;
        k_pack<<<(tot + 255) / 256, 256, 0, stream>>>(W_out, pWo, 1, 22, OUTC);
    }

    // --- input projection: x0 = group_input(...) ---
    k_input_proj<<<NN0 / 16, 256, 0, stream>>>(feats, n_id, node_type, local_idx,
                                               pWin, b_in, x0);

    // --- conv layer 0 ---
    k_gemm<true><<<NN0 / 16, 256, 0, stream>>>(x0, pWc, b_conv, (void*)hh,
                                               NN0, 8, HDIM, HDIM);
    k_zero<<<2048, 256, 0, stream>>>(agg, (long long)NN1 * HDIM);
    k_scatter<<<6144, 256, 0, stream>>>(hh, edge_src0, edge_dst0, etype0,
                                        rel_w, agg, EE0);
    {
        long long thr = (long long)NN1 * HDIM;
        k_combine<<<(unsigned)((thr + 255) / 256), 256, 0, stream>>>(
            agg, hh, x0, n_id, node_type, self_w, x1, NN1);
    }

    // --- conv layer 1 ---
    k_gemm<true><<<NN1 / 16, 256, 0, stream>>>(x1, pWc + (size_t)8 * 4 * 512,
                                               b_conv + HDIM, (void*)hh,
                                               NN1, 8, HDIM, HDIM);
    k_zero<<<2048, 256, 0, stream>>>(agg, (long long)NN2 * HDIM);
    k_scatter<<<6144, 256, 0, stream>>>(hh, edge_src1, edge_dst1, etype1,
                                        rel_w + 8, agg, EE1);
    {
        long long thr = (long long)NN2 * HDIM;
        k_combine<<<(unsigned)((thr + 255) / 256), 256, 0, stream>>>(
            agg, hh, x1, n_id, node_type, self_w + NTYPES, x2, NN2);
    }

    // --- output head: logits = x2 @ W_out + b_out ; log_softmax -> d_out ---
    k_gemm<false><<<(NN2 + 15) / 16, 256, 0, stream>>>(x2, pWo, b_out, (void*)logits,
                                                       NN2, 22, OUTC, OUTP);
    k_lsm<<<NN2, 128, 0, stream>>>(logits, out);
}
